// Mention_guide_56779467653403
// MI455X (gfx1250) — matrix operand order repaired and checked by reading, not compiled, not run on hardware
//
#include <hip/hip_runtime.h>

// ---------------------------------------------------------------------------
// Sizes (fixed by the reference): w=1024, a=128, d_in=d_h=768
// ---------------------------------------------------------------------------
#define W    1024
#define A    128
#define D    768

typedef __attribute__((ext_vector_type(16))) __bf16 v16bf;
typedef __attribute__((ext_vector_type(8)))  float  v8f;

__device__ __forceinline__ unsigned short f2bf(float f) {
    unsigned u = __float_as_uint(f);
    u += 0x7FFFu + ((u >> 16) & 1u);   // round-to-nearest-even
    return (unsigned short)(u >> 16);
}

// ---------------------------------------------------------------------------
// Kernel 1: fold weights.
//   uvc[0..767]   = u[i] = sum_h Wq[i][h] * Wa[h]
//   uvc[768..1535]= v[i] = sum_h Wk[i][h] * Wa[768+h]
//   uvc[1536]     = c    = bq.Wa1 + bk.Wa2 + ba
// One block (256 thr) per output element; block 1536 computes c.
// ---------------------------------------------------------------------------
__global__ void prep_uvc(const float* __restrict__ Wq, const float* __restrict__ Wk,
                         const float* __restrict__ Wa, const float* __restrict__ bq,
                         const float* __restrict__ bk, const float* __restrict__ ba,
                         float* __restrict__ uvc) {
    __shared__ float red[256];
    const int b = blockIdx.x, t = threadIdx.x;
    float p = 0.f;
    if (b < D) {
        const float* r = Wq + (size_t)b * D;
#pragma unroll
        for (int i = 0; i < 3; ++i) p += r[t + 256 * i] * Wa[t + 256 * i];
    } else if (b < 2 * D) {
        const float* r = Wk + (size_t)(b - D) * D;
#pragma unroll
        for (int i = 0; i < 3; ++i) p += r[t + 256 * i] * Wa[D + t + 256 * i];
    } else {
#pragma unroll
        for (int i = 0; i < 3; ++i) {
            int h = t + 256 * i;
            p += bq[h] * Wa[h] + bk[h] * Wa[D + h];
        }
    }
    red[t] = p;
    __syncthreads();
    for (int s = 128; s > 0; s >>= 1) {
        if (t < s) red[t] += red[t + s];
        __syncthreads();
    }
    if (t == 0) {
        if (b < 2 * D) uvc[b] = red[0];
        else           uvc[2 * D] = red[0] + ba[0];
    }
}

// ---------------------------------------------------------------------------
// Kernel 2: WkT_bf16[n][k] = bf16(Wk[k][n]) — contiguous-K layout for WMMA B.
// ---------------------------------------------------------------------------
__global__ void wk_transpose(const float* __restrict__ Wk, unsigned short* __restrict__ wkt) {
    __shared__ float tile[32][33];
    const int nb = blockIdx.x * 32, kb = blockIdx.y * 32;
    const int tx = threadIdx.x, ty = threadIdx.y;        // 32 x 8
#pragma unroll
    for (int i = 0; i < 32; i += 8)
        tile[ty + i][tx] = Wk[(size_t)(kb + ty + i) * D + nb + tx];
    __syncthreads();
#pragma unroll
    for (int i = 0; i < 32; i += 8)
        wkt[(size_t)(nb + ty + i) * D + kb + tx] = f2bf(tile[tx][ty + i]);
}

// ---------------------------------------------------------------------------
// Kernel 3: single streaming pass over key (403 MB -> the roofline limiter).
// One workgroup per w. 8 waves x 16 rows. Lane owns h-slots {j*128+lane*4+q}.
// Per row: float4 loads, dot with v, lrelu+mask, online softmax, p-weighted
// accumulate into VGPR context. Cross-wave merge in LDS. Emits bf16 ctx.
// ---------------------------------------------------------------------------
__global__ void attn_stream(const float* __restrict__ query, const float* __restrict__ key,
                            const int* __restrict__ mask, const float* __restrict__ uvc,
                            unsigned short* __restrict__ ctxbf) {
    __shared__ float red[256];
    __shared__ __align__(16) float ctxPart[8 * D];
    __shared__ float mPart[8], sPart[8], qdotSh;

    const int w = blockIdx.x;
    const int t = threadIdx.x;
    const int lane = t & 31;
    const int wv = t >> 5;

    // qdot = query[w] . u + c (block reduction)
    float qp = 0.f;
#pragma unroll
    for (int i = 0; i < 3; ++i) qp += query[(size_t)w * D + t + 256 * i] * uvc[t + 256 * i];
    red[t] = qp;
    __syncthreads();
    if (t < 32) {
        float s = 0.f;
#pragma unroll
        for (int i = 0; i < 8; ++i) s += red[t + 32 * i];
#pragma unroll
        for (int off = 16; off > 0; off >>= 1) s += __shfl_xor(s, off, 32);
        if (t == 0) qdotSh = s + uvc[2 * D];
    }
    __syncthreads();
    const float qdot = qdotSh;

    // per-lane chunk of v
    float vreg[24];
#pragma unroll
    for (int j = 0; j < 6; ++j) {
        const float4 vv = *(const float4*)(uvc + D + j * 128 + lane * 4);
        vreg[j * 4 + 0] = vv.x; vreg[j * 4 + 1] = vv.y;
        vreg[j * 4 + 2] = vv.z; vreg[j * 4 + 3] = vv.w;
    }

    float m = -__builtin_inff();
    float ssum = 0.f;
    float ctx[24];
#pragma unroll
    for (int i = 0; i < 24; ++i) ctx[i] = 0.f;

    const float* keyw = key + (size_t)w * A * D;
    for (int r = 0; r < 16; ++r) {
        const int a = wv * 16 + r;
        const float* rowp = keyw + (size_t)a * D;
        float kv[24];
        float dot = 0.f;
#pragma unroll
        for (int j = 0; j < 6; ++j) {
            const float4 kk = *(const float4*)(rowp + j * 128 + lane * 4);
            if (r < 15) __builtin_prefetch(rowp + D + j * 128 + lane * 4, 0, 0);
            kv[j * 4 + 0] = kk.x; kv[j * 4 + 1] = kk.y;
            kv[j * 4 + 2] = kk.z; kv[j * 4 + 3] = kk.w;
            dot = fmaf(kk.x, vreg[j * 4 + 0], dot);
            dot = fmaf(kk.y, vreg[j * 4 + 1], dot);
            dot = fmaf(kk.z, vreg[j * 4 + 2], dot);
            dot = fmaf(kk.w, vreg[j * 4 + 3], dot);
        }
#pragma unroll
        for (int off = 16; off > 0; off >>= 1) dot += __shfl_xor(dot, off, 32);

        float sc = qdot + dot;
        sc = sc > 0.f ? sc : 0.01f * sc;                  // leaky_relu
        sc = mask[w * A + a] ? sc : -1e9f;                // masked_fill

        if (sc > m) {                                     // online softmax rescale
            const float scale = __expf(m - sc);
            ssum *= scale;
#pragma unroll
            for (int i = 0; i < 24; ++i) ctx[i] *= scale;
            m = sc;
        }
        const float p = __expf(sc - m);
        ssum += p;
#pragma unroll
        for (int i = 0; i < 24; ++i) ctx[i] = fmaf(p, kv[i], ctx[i]);
    }

    // publish per-wave partials
#pragma unroll
    for (int j = 0; j < 6; ++j) {
        float4 v4 = make_float4(ctx[j * 4 + 0], ctx[j * 4 + 1], ctx[j * 4 + 2], ctx[j * 4 + 3]);
        *(float4*)&ctxPart[wv * D + j * 128 + lane * 4] = v4;
    }
    if (lane == 0) { mPart[wv] = m; sPart[wv] = ssum; }
    __syncthreads();

    // merge 8 partials (log-sum-exp), normalize, emit bf16 ctx row
    float M = mPart[0];
#pragma unroll
    for (int i = 1; i < 8; ++i) M = fmaxf(M, mPart[i]);
    float wgt[8], S = 0.f;
#pragma unroll
    for (int i = 0; i < 8; ++i) { wgt[i] = __expf(mPart[i] - M); S = fmaf(sPart[i], wgt[i], S); }
    const float inv = 1.f / S;
#pragma unroll
    for (int i = 0; i < 3; ++i) {
        const int h = t + 256 * i;
        float acc = 0.f;
#pragma unroll
        for (int k = 0; k < 8; ++k) acc = fmaf(ctxPart[k * D + h], wgt[k], acc);
        ctxbf[(size_t)w * D + h] = f2bf(acc * inv);
    }
}

// ---------------------------------------------------------------------------
// Kernel 4: out = ctx @ Wk + bk  via v_wmma_f32_16x16x32_bf16.
// One 16x16 output tile per wave; K-loop of 24 steps.
//  A (16x32 bf16): lane<16 -> row, K = k0+{0..7,16..23}; lane>=16 -> +8 shift.
//  B (32x16 bf16): lane<16 -> col, K = k0+{0..7,16..23}; lane>=16 -> +8 shift (from WkT rows).
//  C/D: VGPR r -> M = r + 8*half, N = lane&15.
// ---------------------------------------------------------------------------
__global__ void out_gemm(const unsigned short* __restrict__ ctxbf,
                         const unsigned short* __restrict__ wkt,
                         const float* __restrict__ bk, float* __restrict__ out) {
    const int t = threadIdx.x;
    const int lane = t & 31, wv = t >> 5;
    const int tile = blockIdx.x * 8 + wv;       // 64 x 48 = 3072 tiles total
    const int tm = tile / 48, tn = tile % 48;
    const int half = lane >> 4, q = lane & 15;

    const unsigned short* ap = ctxbf + (size_t)(tm * 16 + q) * D;   // A row
    const unsigned short* bp = wkt   + (size_t)(tn * 16 + q) * D;   // B col (K-contig)

    v8f acc = {};
    for (int k0 = 0; k0 < D; k0 += 32) {
        union { v16bf v; uint4 u[2]; } Af, Bf;
        Af.u[0] = *(const uint4*)(ap + k0 + half * 8);
        Af.u[1] = *(const uint4*)(ap + k0 + 16 + half * 8);
        Bf.u[0] = *(const uint4*)(bp + k0 + half * 8);
        Bf.u[1] = *(const uint4*)(bp + k0 + 16 + half * 8);
        acc = __builtin_amdgcn_wmma_f32_16x16x32_bf16(
            false, Af.v, false, Bf.v, (short)0, acc, false, false);
    }
    const float bkc = bk[tn * 16 + q];
#pragma unroll
    for (int r = 0; r < 8; ++r)
        out[(size_t)(tm * 16 + half * 8 + r) * D + tn * 16 + q] = acc[r] + bkc;
}

// ---------------------------------------------------------------------------
// Launch. d_ws layout (bytes):
//   [0, 6148)        uvc: u[768] f32, v[768] f32, c f32
//   [8192, 1581056)  ctx bf16 [1024][768]
//   [1581056, ...)   WkT bf16 [768][768]
// ---------------------------------------------------------------------------
extern "C" void kernel_launch(void* const* d_in, const int* in_sizes, int n_in,
                              void* d_out, int out_size, void* d_ws, size_t ws_size,
                              hipStream_t stream) {
    const float* query = (const float*)d_in[0];
    const float* key   = (const float*)d_in[1];
    const int*   mask  = (const int*)d_in[2];
    const float* Wq    = (const float*)d_in[3];
    const float* bq    = (const float*)d_in[4];
    const float* Wk    = (const float*)d_in[5];
    const float* bk    = (const float*)d_in[6];
    const float* Wa    = (const float*)d_in[7];
    const float* ba    = (const float*)d_in[8];
    float* out = (float*)d_out;

    char* ws = (char*)d_ws;
    float*          uvc   = (float*)ws;
    unsigned short* ctxbf = (unsigned short*)(ws + 8192);
    unsigned short* wkt   = (unsigned short*)(ws + 8192 + (size_t)W * D * 2);

    prep_uvc<<<2 * D + 1, 256, 0, stream>>>(Wq, Wk, Wa, bq, bk, ba, uvc);
    wk_transpose<<<dim3(D / 32, D / 32), dim3(32, 8), 0, stream>>>(Wk, wkt);
    attn_stream<<<W, 256, 0, stream>>>(query, key, mask, uvc, ctxbf);
    out_gemm<<<(W / 16) * (D / 16) / 8, 256, 0, stream>>>(ctxbf, wkt, bk, out);
}
